// Encoder_86852828659979
// MI455X (gfx1250) — matrix operand order repaired and checked by reading, not compiled, not run on hardware
//
#include <hip/hip_runtime.h>
#include <math.h>

// ---------------------------------------------------------------------------
// Problem constants: S=2048, D=512, H=8, HID=2048
// ---------------------------------------------------------------------------
#define S_DIM   2048
#define D_DIM   512
#define H_NUM   8
#define HID_DIM 2048
#define LN_EPS  1e-5f

typedef __attribute__((ext_vector_type(16))) __bf16 v16bf;
typedef __attribute__((ext_vector_type(8)))  float  v8f;

__device__ __forceinline__ unsigned short f2bf(float f) {
    // round-to-nearest-even f32 -> bf16
    unsigned int u = __float_as_uint(f);
    unsigned int r = u + 0x7FFFu + ((u >> 16) & 1u);
    return (unsigned short)(r >> 16);
}

// Generic pointer to an LDS object: low 32 bits are the LDS byte address
// (flat LDS aperture truncates to addr[31:0], matching async VDST semantics).
__device__ __forceinline__ unsigned lds_addr32(const void* p) {
    return (unsigned)(unsigned long long)p;
}

// CDNA5 async global->LDS copy (16B per lane), tracked with ASYNCcnt.
__device__ __forceinline__ void async_copy_b128(unsigned dst_lds, const void* src) {
    asm volatile("global_load_async_to_lds_b128 %0, %1, off"
                 :: "v"(dst_lds), "v"(src) : "memory");
}
__device__ __forceinline__ void wait_async_le8() {
    asm volatile("s_wait_asynccnt 8" ::: "memory");
}
__device__ __forceinline__ void wait_async_0() {
    asm volatile("s_wait_asynccnt 0" ::: "memory");
}

// ---------------------------------------------------------------------------
// Uniform bf16 WMMA GEMM:  C[M,N] = A[M,K] @ B^T  (B stored [N,K] row-major)
// Both operands bf16 in global memory. f32 accumulate.
// Block tile 128x128, K-tile 64 (2 WMMA k-steps), 8 waves, wave tile 32x64.
// Double-buffered LDS fed by global_load_async_to_lds_b128.
// OUTMODE: 0 = f32 C[m*ldc+n], 1 = bf16 C[m*ldc+n], 2 = bf16 C[n*ldc+m]
// ---------------------------------------------------------------------------
constexpr int BM = 128, BN = 128, BK = 64;
constexpr int LDS_STR = BK + 8;   // 72 elems = 144B rows: 16B-aligned, bank-staggered

template<bool BIAS, bool RELU, int OUTMODE>
__global__ __launch_bounds__(256)
void gemm_bf16_wmma(const unsigned short* __restrict__ Ag,
                    const unsigned short* __restrict__ Bg,
                    const float* __restrict__ biasg, void* __restrict__ Cg,
                    int M, int N, int K, int lda, int ldb, int ldc,
                    long long sA, long long sB, long long sBias, long long sC)
{
    __shared__ __align__(16) unsigned short shA[2][BM * LDS_STR];
    __shared__ __align__(16) unsigned short shB[2][BN * LDS_STR];

    const int bz = blockIdx.z;
    const unsigned short* A = Ag + (long long)bz * sA;
    const unsigned short* B = Bg + (long long)bz * sB;

    const int m0 = blockIdx.y * BM;
    const int n0 = blockIdx.x * BN;

    const int tid  = threadIdx.x;
    const int lane = tid & 31;
    const int wave = tid >> 5;
    const int wm   = wave & 3;     // 32-row slab
    const int wn   = wave >> 2;    // 64-col slab
    const int lh   = lane >> 4;    // lane half
    const int lr   = lane & 15;

    // staging geometry: thread -> (row, 8-elem chunk), 4 row-iterations
    const int srow = tid >> 3;          // 0..31
    const int sc8  = (tid & 7) * 8;     // element offset of 16B chunk

    v8f acc[2][4];
    #pragma unroll
    for (int i = 0; i < 2; ++i)
        #pragma unroll
        for (int j = 0; j < 4; ++j)
            acc[i][j] = (v8f){0.f,0.f,0.f,0.f,0.f,0.f,0.f,0.f};

    // ---- issue stage 0 ----
    {
        #pragma unroll
        for (int it = 0; it < 4; ++it) {
            int row = srow + it * 32;
            async_copy_b128(lds_addr32(&shA[0][row * LDS_STR + sc8]),
                            A + (long long)(m0 + row) * lda + sc8);
            async_copy_b128(lds_addr32(&shB[0][row * LDS_STR + sc8]),
                            B + (long long)(n0 + row) * ldb + sc8);
        }
    }

    const int nstage = K / BK;
    for (int st = 0; st < nstage; ++st) {
        const int cur = st & 1;
        if (st + 1 < nstage) {
            const int k0 = (st + 1) * BK;
            #pragma unroll
            for (int it = 0; it < 4; ++it) {
                int row = srow + it * 32;
                async_copy_b128(lds_addr32(&shA[cur ^ 1][row * LDS_STR + sc8]),
                                A + (long long)(m0 + row) * lda + k0 + sc8);
                async_copy_b128(lds_addr32(&shB[cur ^ 1][row * LDS_STR + sc8]),
                                B + (long long)(n0 + row) * ldb + k0 + sc8);
            }
            wait_async_le8();   // stage `cur` complete; next stage may be in flight
        } else {
            wait_async_0();
        }
        __syncthreads();

        const unsigned short* bufA = &shA[cur][0];
        const unsigned short* bufB = &shB[cur][0];
        #pragma unroll
        for (int s = 0; s < 2; ++s) {          // two 16x16x32 k-steps per stage
            v16bf af[2], bf[4];
            #pragma unroll
            for (int i = 0; i < 2; ++i) {
                const unsigned short* base = bufA + (wm * 32 + i * 16 + lr) * LDS_STR + s * 32;
                union { v16bf v; uint4 q[2]; } t;
                t.q[0] = *(const uint4*)(base + lh * 8);        // A: K = lh*8 .. +7
                t.q[1] = *(const uint4*)(base + 16 + lh * 8);   //    K = 16+lh*8 .. +7
                af[i] = t.v;
            }
            #pragma unroll
            for (int j = 0; j < 4; ++j) {
                const unsigned short* base = bufB + (wn * 64 + j * 16 + lr) * LDS_STR + s * 32;
                union { v16bf v; uint4 q[2]; } t;
                t.q[0] = *(const uint4*)(base + lh * 8);        // B: K = lh*8 .. +7
                t.q[1] = *(const uint4*)(base + 16 + lh * 8);   //    K = 16+lh*8 .. +7
                bf[j] = t.v;
            }
            #pragma unroll
            for (int i = 0; i < 2; ++i)
                #pragma unroll
                for (int j = 0; j < 4; ++j)
                    acc[i][j] = __builtin_amdgcn_wmma_f32_16x16x32_bf16(
                        false, af[i], false, bf[j], (short)0, acc[i][j], false, false);
        }
        __syncthreads();
    }

    // ---- epilogue: C/D layout m = e + 8*lh, n = lr ----
    const float* bias = BIAS ? (biasg + (long long)bz * sBias) : nullptr;
    #pragma unroll
    for (int i = 0; i < 2; ++i) {
        int mbase = m0 + wm * 32 + i * 16 + lh * 8;
        #pragma unroll
        for (int j = 0; j < 4; ++j) {
            int n = n0 + wn * 64 + j * 16 + lr;
            float bv = BIAS ? bias[n] : 0.f;
            #pragma unroll
            for (int e = 0; e < 8; ++e) {
                float v = acc[i][j][e] + bv;
                if (RELU) v = fmaxf(v, 0.f);
                if constexpr (OUTMODE == 0) {
                    float* C = (float*)Cg + (long long)bz * sC;
                    C[(long long)(mbase + e) * ldc + n] = v;
                } else if constexpr (OUTMODE == 1) {
                    unsigned short* C = (unsigned short*)Cg + (long long)bz * sC;
                    C[(long long)(mbase + e) * ldc + n] = f2bf(v);
                } else {
                    unsigned short* C = (unsigned short*)Cg + (long long)bz * sC;
                    C[(long long)n * ldc + (mbase + e)] = f2bf(v);
                }
            }
        }
    }
}

// ---------------------------------------------------------------------------
// f32 -> bf16 elementwise convert (n multiple of 1024)
// ---------------------------------------------------------------------------
__global__ __launch_bounds__(256)
void convert_bf16_kernel(const float* __restrict__ in, unsigned short* __restrict__ out,
                         long long n)
{
    long long i = ((long long)blockIdx.x * 256 + threadIdx.x) * 4;
    if (i < n) {
        float4 f = *(const float4*)(in + i);
        ushort4 o;
        o.x = f2bf(f.x); o.y = f2bf(f.y); o.z = f2bf(f.z); o.w = f2bf(f.w);
        *(ushort4*)(out + i) = o;
    }
}

// ---------------------------------------------------------------------------
// f32 [R,C] -> bf16 [C,R] transpose-convert (R,C multiples of 32), batched
// ---------------------------------------------------------------------------
__global__ __launch_bounds__(256)
void transpose_bf16_kernel(const float* __restrict__ in, unsigned short* __restrict__ out,
                           int R, int C, long long sIn, long long sOut)
{
    __shared__ float tile[32][33];
    const float* A = in + (long long)blockIdx.z * sIn;
    unsigned short* O = out + (long long)blockIdx.z * sOut;
    int r0 = blockIdx.y * 32, c0 = blockIdx.x * 32;
    int tx = threadIdx.x & 31, ty = threadIdx.x >> 5;
    #pragma unroll
    for (int i = 0; i < 32; i += 8)
        tile[ty + i][tx] = A[(long long)(r0 + ty + i) * C + c0 + tx];
    __syncthreads();
    #pragma unroll
    for (int i = 0; i < 32; i += 8)
        O[(long long)(c0 + ty + i) * R + r0 + tx] = f2bf(tile[tx][ty + i]);
}

// ---------------------------------------------------------------------------
// Row softmax (reference quirk): p = softmax(row) * post_scale, bf16 out
// ---------------------------------------------------------------------------
__global__ __launch_bounds__(256)
void softmax_kernel(float* __restrict__ sc, unsigned short* __restrict__ pout,
                    int N, float post_scale)
{
    long long row = blockIdx.x;
    float* p = sc + row * (long long)N;
    unsigned short* po = pout + row * (long long)N;
    int tid = threadIdx.x;
    __shared__ float red[8];

    float m = -3.0e38f;
    for (int i = tid; i < N; i += 256) m = fmaxf(m, p[i]);
    #pragma unroll
    for (int o = 16; o > 0; o >>= 1) m = fmaxf(m, __shfl_xor(m, o, 32));
    if ((tid & 31) == 0) red[tid >> 5] = m;
    __syncthreads();
    float bm = red[0];
    #pragma unroll
    for (int i = 1; i < 8; ++i) bm = fmaxf(bm, red[i]);
    __syncthreads();

    float s = 0.f;
    for (int i = tid; i < N; i += 256) { float e = __expf(p[i] - bm); p[i] = e; s += e; }
    #pragma unroll
    for (int o = 16; o > 0; o >>= 1) s += __shfl_xor(s, o, 32);
    if ((tid & 31) == 0) red[tid >> 5] = s;
    __syncthreads();
    float ts = 0.f;
    #pragma unroll
    for (int i = 0; i < 8; ++i) ts += red[i];

    float scale = post_scale / ts;
    for (int i = tid; i < N; i += 256) po[i] = f2bf(p[i] * scale);
}

// ---------------------------------------------------------------------------
// out = LayerNorm(a + b); optionally also emit bf16 copy. D fixed 512.
// ---------------------------------------------------------------------------
template<bool DUAL>
__global__ __launch_bounds__(256)
void add_ln_kernel(const float* __restrict__ a, const float* __restrict__ b,
                   const float* __restrict__ g, const float* __restrict__ bb,
                   float* __restrict__ out, unsigned short* __restrict__ outb)
{
    long long row = blockIdx.x;
    const float* pa = a + row * D_DIM;
    const float* pb = b + row * D_DIM;
    int tid = threadIdx.x;
    __shared__ float red[8];

    float x0 = pa[tid]       + pb[tid];
    float x1 = pa[tid + 256] + pb[tid + 256];

    float s = x0 + x1;
    #pragma unroll
    for (int o = 16; o > 0; o >>= 1) s += __shfl_xor(s, o, 32);
    if ((tid & 31) == 0) red[tid >> 5] = s;
    __syncthreads();
    float tot = 0.f;
    #pragma unroll
    for (int i = 0; i < 8; ++i) tot += red[i];
    float mean = tot * (1.0f / D_DIM);
    __syncthreads();

    float d0 = x0 - mean, d1 = x1 - mean;
    float vs = d0 * d0 + d1 * d1;
    #pragma unroll
    for (int o = 16; o > 0; o >>= 1) vs += __shfl_xor(vs, o, 32);
    if ((tid & 31) == 0) red[tid >> 5] = vs;
    __syncthreads();
    float vtot = 0.f;
    #pragma unroll
    for (int i = 0; i < 8; ++i) vtot += red[i];
    float inv = rsqrtf(vtot * (1.0f / D_DIM) + LN_EPS);

    float y0 = d0 * inv * g[tid]       + bb[tid];
    float y1 = d1 * inv * g[tid + 256] + bb[tid + 256];
    out[row * D_DIM + tid]       = y0;
    out[row * D_DIM + tid + 256] = y1;
    if (DUAL) {
        outb[row * D_DIM + tid]       = f2bf(y0);
        outb[row * D_DIM + tid + 256] = f2bf(y1);
    }
}

// ---------------------------------------------------------------------------
// Launcher
// ---------------------------------------------------------------------------
extern "C" void kernel_launch(void* const* d_in, const int* in_sizes, int n_in,
                              void* d_out, int out_size, void* d_ws, size_t ws_size,
                              hipStream_t stream)
{
    (void)in_sizes; (void)n_in; (void)out_size; (void)ws_size;

    const float* x    = (const float*)d_in[0];
    const float* Wq   = (const float*)d_in[1];
    const float* bq   = (const float*)d_in[2];
    const float* Wk   = (const float*)d_in[3];
    const float* bk   = (const float*)d_in[4];
    const float* Wv   = (const float*)d_in[5];
    const float* bv   = (const float*)d_in[6];
    const float* Wo   = (const float*)d_in[7];
    const float* bo   = (const float*)d_in[8];
    const float* ln1g = (const float*)d_in[9];
    const float* ln1b = (const float*)d_in[10];
    const float* ln2g = (const float*)d_in[11];
    const float* ln2b = (const float*)d_in[12];
    const float* W1   = (const float*)d_in[13];
    const float* b1   = (const float*)d_in[14];
    const float* W2   = (const float*)d_in[15];
    const float* b2   = (const float*)d_in[16];
    const float* W3   = (const float*)d_in[17];
    const float* b3   = (const float*)d_in[18];
    const float* W4   = (const float*)d_in[19];
    const float* b4   = (const float*)d_in[20];
    float* out = (float*)d_out;

    typedef unsigned short u16;
    const long long SD  = (long long)S_DIM * D_DIM;          // 1M
    const long long HSD = (long long)H_NUM * SD;             // 8M
    const long long SS  = (long long)S_DIM * S_DIM;          // 4M
    const long long HSS = (long long)H_NUM * SS;             // 32M
    const long long DD  = (long long)D_DIM * D_DIM;
    const long long HD  = (long long)H_NUM * D_DIM;          // 4096

    char* ws = (char*)d_ws;
    u16* xb   = (u16*)ws; ws += SD * 2;
    u16* WqT  = (u16*)ws; ws += H_NUM * DD * 2;
    u16* WkT  = (u16*)ws; ws += H_NUM * DD * 2;
    u16* WvT  = (u16*)ws; ws += H_NUM * DD * 2;
    u16* WoT  = (u16*)ws; ws += HD * D_DIM * 2;
    u16* W1T  = (u16*)ws; ws += (long long)D_DIM * HID_DIM * 2;
    u16* W2T  = (u16*)ws; ws += (long long)HID_DIM * HID_DIM * 2;
    u16* W3T  = (u16*)ws; ws += (long long)HID_DIM * HID_DIM * 2;
    u16* W4T  = (u16*)ws; ws += (long long)HID_DIM * D_DIM * 2;
    u16* qb   = (u16*)ws; ws += HSD * 2;
    u16* kb   = (u16*)ws; ws += HSD * 2;
    u16* vT   = (u16*)ws; ws += HSD * 2;   // per-head [D, S]
    u16* pb   = (u16*)ws; ws += HSS * 2;
    u16* zcat = (u16*)ws; ws += HSD * 2;
    u16* zlnb = (u16*)ws; ws += SD * 2;
    float* sc  = (float*)ws; ws += HSS * 4;
    float* zo  = (float*)ws; ws += SD * 4;
    float* zln = (float*)ws; ws += SD * 4;
    // FFN hidden activations reuse the (dead) scores region
    u16* h1 = (u16*)sc;
    u16* h2 = h1 + (long long)S_DIM * HID_DIM;
    u16* h3 = h2 + (long long)S_DIM * HID_DIM;

    dim3 blk(256);

    // ---- precision staging: convert / transpose weights & x to bf16 ----
    convert_bf16_kernel<<<dim3((unsigned)(SD / 1024)), blk, 0, stream>>>(x, xb, SD);
    transpose_bf16_kernel<<<dim3(D_DIM/32, D_DIM/32, H_NUM), blk, 0, stream>>>(Wq, WqT, D_DIM, D_DIM, DD, DD);
    transpose_bf16_kernel<<<dim3(D_DIM/32, D_DIM/32, H_NUM), blk, 0, stream>>>(Wk, WkT, D_DIM, D_DIM, DD, DD);
    transpose_bf16_kernel<<<dim3(D_DIM/32, D_DIM/32, H_NUM), blk, 0, stream>>>(Wv, WvT, D_DIM, D_DIM, DD, DD);
    transpose_bf16_kernel<<<dim3(D_DIM/32, (unsigned)(HD/32), 1), blk, 0, stream>>>(Wo, WoT, HD, D_DIM, 0, 0);
    transpose_bf16_kernel<<<dim3(HID_DIM/32, D_DIM/32, 1), blk, 0, stream>>>(W1, W1T, D_DIM, HID_DIM, 0, 0);
    transpose_bf16_kernel<<<dim3(HID_DIM/32, HID_DIM/32, 1), blk, 0, stream>>>(W2, W2T, HID_DIM, HID_DIM, 0, 0);
    transpose_bf16_kernel<<<dim3(HID_DIM/32, HID_DIM/32, 1), blk, 0, stream>>>(W3, W3T, HID_DIM, HID_DIM, 0, 0);
    transpose_bf16_kernel<<<dim3(D_DIM/32, HID_DIM/32, 1), blk, 0, stream>>>(W4, W4T, HID_DIM, D_DIM, 0, 0);

    // ---- QKV: q/k = x @ W + b (bf16 out); v written transposed [D,S] ----
    dim3 g_qkv(D_DIM / BN, S_DIM / BM, H_NUM);
    gemm_bf16_wmma<true, false, 1><<<g_qkv, blk, 0, stream>>>(
        xb, WqT, bq, qb, S_DIM, D_DIM, D_DIM, D_DIM, D_DIM, D_DIM, 0, DD, D_DIM, SD);
    gemm_bf16_wmma<true, false, 1><<<g_qkv, blk, 0, stream>>>(
        xb, WkT, bk, kb, S_DIM, D_DIM, D_DIM, D_DIM, D_DIM, D_DIM, 0, DD, D_DIM, SD);
    gemm_bf16_wmma<true, false, 2><<<g_qkv, blk, 0, stream>>>(
        xb, WvT, bv, vT, S_DIM, D_DIM, D_DIM, D_DIM, D_DIM, S_DIM, 0, DD, D_DIM, SD);

    // ---- scores[h] = q[h] @ k[h]^T  (f32 out) ----
    dim3 g_sc(S_DIM / BN, S_DIM / BM, H_NUM);
    gemm_bf16_wmma<false, false, 0><<<g_sc, blk, 0, stream>>>(
        qb, kb, nullptr, sc, S_DIM, S_DIM, D_DIM, D_DIM, D_DIM, S_DIM, SD, SD, 0, SS);

    // ---- p = softmax(scores) / sqrt(D)  (bf16 out) ----
    softmax_kernel<<<dim3(H_NUM * S_DIM), blk, 0, stream>>>(
        sc, pb, S_DIM, 1.0f / sqrtf((float)D_DIM));

    // ---- z[h] = p[h] @ v[h]: B = vT[h] ([D,S]); write into concat [S,H*D] ----
    dim3 g_pv(D_DIM / BN, S_DIM / BM, H_NUM);
    gemm_bf16_wmma<false, false, 1><<<g_pv, blk, 0, stream>>>(
        pb, vT, nullptr, zcat, S_DIM, D_DIM, S_DIM, S_DIM, S_DIM, (int)HD,
        SS, SD, 0, (long long)D_DIM);

    // ---- zo = zcat @ Wo + bo  (f32) ----
    dim3 g_wo(D_DIM / BN, S_DIM / BM, 1);
    gemm_bf16_wmma<true, false, 0><<<g_wo, blk, 0, stream>>>(
        zcat, WoT, bo, zo, S_DIM, D_DIM, (int)HD, (int)HD, (int)HD, D_DIM, 0, 0, 0, 0);

    // ---- zln = LN(x + zo), plus bf16 copy for FFN input ----
    add_ln_kernel<true><<<dim3(S_DIM), blk, 0, stream>>>(x, zo, ln1g, ln1b, zln, zlnb);

    // ---- FFN (bf16 activations, ReLU fused) ----
    dim3 g_f(HID_DIM / BN, S_DIM / BM, 1);
    gemm_bf16_wmma<true, true, 1><<<g_f, blk, 0, stream>>>(
        zlnb, W1T, b1, h1, S_DIM, HID_DIM, D_DIM, D_DIM, D_DIM, HID_DIM, 0, 0, 0, 0);
    gemm_bf16_wmma<true, true, 1><<<g_f, blk, 0, stream>>>(
        h1, W2T, b2, h2, S_DIM, HID_DIM, HID_DIM, HID_DIM, HID_DIM, HID_DIM, 0, 0, 0, 0);
    gemm_bf16_wmma<true, true, 1><<<g_f, blk, 0, stream>>>(
        h2, W3T, b3, h3, S_DIM, HID_DIM, HID_DIM, HID_DIM, HID_DIM, HID_DIM, 0, 0, 0, 0);
    gemm_bf16_wmma<true, false, 0><<<g_wo, blk, 0, stream>>>(
        h3, W4T, b4, zo, S_DIM, D_DIM, HID_DIM, HID_DIM, HID_DIM, D_DIM, 0, 0, 0, 0);

    // ---- out = LN(zln + z1) ----
    add_ln_kernel<false><<<dim3(S_DIM), blk, 0, stream>>>(
        zln, zo, ln2g, ln2b, out, nullptr);
}
